// GCN_rec_77464030151238
// MI455X (gfx1250) — matrix operand order repaired and checked by reading, not compiled, not run on hardware
//
#include <hip/hip_runtime.h>
#include <hip/hip_bf16.h>

// ---------------------------------------------------------------------------
// Types for CDNA5 WMMA (wave32, 16x16x32 bf16 -> f32)
// ---------------------------------------------------------------------------
typedef __attribute__((ext_vector_type(16))) __bf16 v16bf;
typedef __attribute__((ext_vector_type(8)))  __bf16 v8bf;
typedef __attribute__((ext_vector_type(8)))  float  v8f;
typedef __attribute__((ext_vector_type(4)))  float  v4f;

union AFrag { v16bf v; v8bf h[2]; };

#define EPI_NONE      0
#define EPI_BIAS      1
#define EPI_BIAS_RELU 2
#define EPI_SCALE     3

// ---------------------------------------------------------------------------
// Generic tiled GEMM: C[M,Nc] = A[M,K] * X  (+ epilogue)
//   A fp32 row-major (lda, per-z stride aZ). X fp32: XT=false -> [K,Nc] (ldx);
//   XT=true -> stored [Nc,K] (ldx), used transposed. C fp32 (ldc, cZ).
// Block: 256 thr = 8 waves; tile 64 rows x BN=WC*8 cols; wave owns WC cols.
// fp32 -> bf16 conversion into LDS; register double-buffered global staging;
// WMMA bf16 16x16x32 with f32 accumulation.
// ---------------------------------------------------------------------------
template<int EPI, bool XT, int WC>
__global__ __launch_bounds__(256)
void gemm_wmma(const float* __restrict__ A, long lda, long aZ,
               const float* __restrict__ X, long ldx, long xZ,
               float* __restrict__ C, long ldc, long cZ,
               const float* __restrict__ bias,
               int M, int K, int Nc, float escale)
{
    constexpr int BN  = WC * 8;     // columns covered by the block
    constexpr int NB  = WC / 16;    // B fragments per wave
    constexpr int PAD = 40;         // padded row (bf16) -> 80B stride, conflict-free
    constexpr int XSZ = BN / 8;     // staged X floats per thread (16 or 32)

    __shared__ alignas(16) __bf16 Asl[2][64 * PAD];
    __shared__ alignas(16) __bf16 Xsl[2][BN * PAD];

    const int z = blockIdx.z;
    A += (long)z * aZ;  X += (long)z * xZ;  C += (long)z * cZ;

    const int rowBase = blockIdx.x * 64;
    const int colBase = blockIdx.y * BN;
    const int tid  = threadIdx.x;
    const int wave = tid >> 5;
    const int lane = tid & 31;
    const int l16  = lane & 15;
    const int lhalf = lane >> 4;
    const int a0 = lhalf ? 8 : 0;       // A-frag K base (ISA 16x16x32 layout)
    const int kb = lhalf ? 8 : 0;       // B-frag K base

    // ---- staging helpers (loads keep fp32 in regs; convert at commit) -----
    const int ar = tid >> 2;            // A: row 0..63
    const int ac = (tid & 3) * 8;       // A: k offset 0,8,16,24
    const long agr = rowBase + ar;

    auto stage = [&](int k0, float* aR, float* xR) {
        // A tile chunk: 8 floats
        const float* asrc = A + agr * lda + k0 + ac;
        if (agr < (long)M && (k0 + 32) <= K) {
            v4f f0 = *(const v4f*)asrc;
            v4f f1 = *(const v4f*)(asrc + 4);
            #pragma unroll
            for (int i = 0; i < 4; ++i) { aR[i] = f0[i]; aR[4 + i] = f1[i]; }
            if (k0 + 64 <= K) __builtin_prefetch(asrc + 32, 0, 1); // global_prefetch_b8
        } else {
            #pragma unroll
            for (int i = 0; i < 8; ++i)
                aR[i] = (agr < (long)M && (k0 + ac + i) < K) ? asrc[i] : 0.0f;
        }
        // X tile
        if constexpr (!XT) {
            #pragma unroll
            for (int u = 0; u < BN / 32; ++u) {           // 4-wide chunks along n
                const int idx = tid + u * 256;
                const int kk  = idx / (BN / 4);
                const int n0  = (idx % (BN / 4)) * 4;
                const long gk = k0 + kk;
                const long gn = colBase + n0;
                if ((colBase + n0 + 4) <= Nc && (k0 + 32) <= K) {
                    v4f f = *(const v4f*)(X + gk * ldx + gn);
                    #pragma unroll
                    for (int i = 0; i < 4; ++i) xR[u * 4 + i] = f[i];
                } else {
                    #pragma unroll
                    for (int i = 0; i < 4; ++i)
                        xR[u * 4 + i] = ((gn + i) < (long)Nc && gk < (long)K)
                                        ? X[gk * ldx + gn + i] : 0.0f;
                }
            }
        } else {
            #pragma unroll
            for (int u = 0; u < BN / 128; ++u) {          // 16 contiguous k per unit
                const int n   = (tid >> 1) + u * 128;
                const int kk0 = (tid & 1) * 16;
                const long gn = colBase + n;
                const float* xp = X + gn * ldx + k0 + kk0;
                if (gn < (long)Nc && (k0 + 32) <= K) {
                    #pragma unroll
                    for (int q = 0; q < 4; ++q) {
                        v4f f = *(const v4f*)(xp + q * 4);
                        #pragma unroll
                        for (int i = 0; i < 4; ++i) xR[u * 16 + q * 4 + i] = f[i];
                    }
                } else {
                    #pragma unroll
                    for (int i = 0; i < 16; ++i)
                        xR[u * 16 + i] = (gn < (long)Nc && (k0 + kk0 + i) < K)
                                         ? xp[i] : 0.0f;
                }
            }
        }
    };

    auto commit = [&](int buf, const float* aR, const float* xR) {
        v8bf av;
        #pragma unroll
        for (int i = 0; i < 8; ++i) av[i] = (__bf16)aR[i];
        *(v8bf*)(&Asl[buf][ar * PAD + ac]) = av;          // one ds_store_b128
        if constexpr (!XT) {
            #pragma unroll
            for (int u = 0; u < BN / 32; ++u) {
                const int idx = tid + u * 256;
                const int kk  = idx / (BN / 4);
                const int n0  = (idx % (BN / 4)) * 4;
                #pragma unroll
                for (int i = 0; i < 4; ++i)
                    Xsl[buf][(n0 + i) * PAD + kk] = (__bf16)xR[u * 4 + i];
            }
        } else {
            #pragma unroll
            for (int u = 0; u < BN / 128; ++u) {
                const int n   = (tid >> 1) + u * 128;
                const int kk0 = (tid & 1) * 16;
                v8bf lo, hi;
                #pragma unroll
                for (int i = 0; i < 8; ++i) {
                    lo[i] = (__bf16)xR[u * 16 + i];
                    hi[i] = (__bf16)xR[u * 16 + 8 + i];
                }
                *(v8bf*)(&Xsl[buf][n * PAD + kk0])     = lo;
                *(v8bf*)(&Xsl[buf][n * PAD + kk0 + 8]) = hi;
            }
        }
    };

    v8f acc[4][NB];
    #pragma unroll
    for (int r = 0; r < 4; ++r)
        #pragma unroll
        for (int b = 0; b < NB; ++b)
            #pragma unroll
            for (int j = 0; j < 8; ++j)
                acc[r][b][j] = 0.0f;

    // prologue: stage + commit tile 0
    {
        float aR[8], xR[XSZ];
        stage(0, aR, xR);
        commit(0, aR, xR);
    }

    int cur = 0;
    for (int k0 = 0; k0 < K; k0 += 32) {
        __syncthreads();                       // buf[cur] visible to all waves
        const bool hasNext = (k0 + 32) < K;
        float aN[8], xN[XSZ];
        if (hasNext) stage(k0 + 32, aN, xN);   // global loads in flight

        // ---- compute on buf[cur] -----------------------------------------
        const __bf16* Ab = Asl[cur];
        const __bf16* Xb = Xsl[cur];
        AFrag bf[NB];
        #pragma unroll
        for (int b = 0; b < NB; ++b) {
            const int n = wave * WC + b * 16 + l16;
            bf[b].h[0] = *(const v8bf*)(Xb + n * PAD + kb);
            bf[b].h[1] = *(const v8bf*)(Xb + n * PAD + kb + 16);
        }
        #pragma unroll
        for (int r = 0; r < 4; ++r) {
            const int m = r * 16 + l16;
            AFrag af;
            af.h[0] = *(const v8bf*)(Ab + m * PAD + a0);
            af.h[1] = *(const v8bf*)(Ab + m * PAD + a0 + 16);
            #pragma unroll
            for (int b = 0; b < NB; ++b)
                acc[r][b] = __builtin_amdgcn_wmma_f32_16x16x32_bf16(
                    false, af.v, false, bf[b].v, (short)0, acc[r][b], false, false);
        }

        if (hasNext) commit(cur ^ 1, aN, xN);  // waits loads here, after WMMA
        cur ^= 1;
    }

    // ---- epilogue: VGPR j -> row = 16r + j + 8*lhalf, col = l16 ------------
    #pragma unroll
    for (int r = 0; r < 4; ++r) {
        #pragma unroll
        for (int b = 0; b < NB; ++b) {
            const int col = colBase + wave * WC + b * 16 + l16;
            if (col >= Nc) continue;
            float bv = 0.0f;
            if (EPI == EPI_BIAS || EPI == EPI_BIAS_RELU) bv = bias[col];
            #pragma unroll
            for (int j = 0; j < 8; ++j) {
                const int row = rowBase + r * 16 + lhalf * 8 + j;
                if (row < M) {
                    float v = acc[r][b][j];
                    if (EPI == EPI_BIAS || EPI == EPI_BIAS_RELU) v += bv;
                    if (EPI == EPI_BIAS_RELU) v = fmaxf(v, 0.0f);
                    if (EPI == EPI_SCALE) v *= escale;
                    C[(long)row * ldc + col] = v;
                }
            }
        }
    }
    (void)escale;
}

// ---------------------------------------------------------------------------
// Small kernels
// ---------------------------------------------------------------------------
__global__ void ew_mul(const float* __restrict__ a, const float* __restrict__ b,
                       float* __restrict__ c, int n) {
    int i = blockIdx.x * 256 + threadIdx.x;
    if (i < n) c[i] = a[i] * b[i];
}

__global__ void gather_rows128(const float* __restrict__ x, const int* __restrict__ idx,
                               float* __restrict__ out, int nrow) {
    int t = blockIdx.x * 256 + threadIdx.x;
    if (t >= nrow * 128) return;
    int d = t & 127, r = t >> 7;
    out[t] = x[(long)idx[r] * 128 + d];
}

// cited[n][0..255] = concat(item[n], nf[n]); rows staged through LDS with the
// CDNA5 async global->LDS engine (ASYNCcnt).
__global__ __launch_bounds__(256)
void concat_cited_async(const float* __restrict__ item, const float* __restrict__ nf,
                        float* __restrict__ cited, int n) {
    __shared__ alignas(16) float buf[16 * 256];          // 16 rows x 256 f32
    const int row0 = blockIdx.x * 16;
    const int tid  = threadIdx.x;
    #pragma unroll
    for (int u = 0; u < 2; ++u) {
        const int c   = tid + u * 256;                   // 16B chunk id 0..511
        const int r   = c >> 5;                          // 32 chunks per row
        const int off = (c & 31) * 4;                    // float offset in row
        const long g  = row0 + r;
        if (g < n) {
            const float* s0 = item + g * 128 + off;
            const float* s1 = nf   + g * 128 + off;
            unsigned l0 = (unsigned)(size_t)(const void*)&buf[r * 256 + off];
            unsigned l1 = (unsigned)(size_t)(const void*)&buf[r * 256 + 128 + off];
            asm volatile("global_load_async_to_lds_b128 %0, %1, off"
                         :: "v"(l0), "v"(s0) : "memory");
            asm volatile("global_load_async_to_lds_b128 %0, %1, off"
                         :: "v"(l1), "v"(s1) : "memory");
        }
    }
    asm volatile("s_wait_asynccnt 0x0" ::: "memory");
    __syncthreads();
    #pragma unroll
    for (int u = 0; u < 4; ++u) {
        const int c   = tid + u * 256;                   // 16B chunk id 0..1023
        const int r   = c >> 6;                          // 64 chunks per row
        const int off = (c & 63) * 4;
        const long g  = row0 + r;
        if (g < n)
            *(v4f*)(cited + g * 256 + off) = *(const v4f*)(buf + r * 256 + off);
    }
}

__global__ __launch_bounds__(256)
void softmax512(float* __restrict__ S, int rows) {
    int row  = blockIdx.x * 8 + (threadIdx.x >> 5);
    int lane = threadIdx.x & 31;
    if (row >= rows) return;
    float* p = S + (long)row * 512;
    float v[16];
    float mx = -3.4e38f;
    #pragma unroll
    for (int i = 0; i < 16; ++i) { v[i] = p[lane + 32 * i]; mx = fmaxf(mx, v[i]); }
    #pragma unroll
    for (int o = 16; o > 0; o >>= 1) mx = fmaxf(mx, __shfl_xor(mx, o, 32));
    float sum = 0.0f;
    #pragma unroll
    for (int i = 0; i < 16; ++i) { v[i] = __expf(v[i] - mx); sum += v[i]; }
    #pragma unroll
    for (int o = 16; o > 0; o >>= 1) sum += __shfl_xor(sum, o, 32);
    float inv = 1.0f / sum;
    #pragma unroll
    for (int i = 0; i < 16; ++i) p[lane + 32 * i] = v[i] * inv;
}

__global__ __launch_bounds__(256)
void add_ln128(const float* __restrict__ a, const float* __restrict__ o,
               const float* __restrict__ s, const float* __restrict__ bi,
               float* __restrict__ out, int rows) {
    int row  = blockIdx.x * 8 + (threadIdx.x >> 5);
    int lane = threadIdx.x & 31;
    if (row >= rows) return;
    long base = (long)row * 128;
    float v[4];
    float m = 0.0f;
    #pragma unroll
    for (int i = 0; i < 4; ++i) {
        int c = lane + 32 * i;
        v[i] = a[base + c] + o[base + c];
        m += v[i];
    }
    #pragma unroll
    for (int off = 16; off > 0; off >>= 1) m += __shfl_xor(m, off, 32);
    m *= (1.0f / 128.0f);
    float var = 0.0f;
    #pragma unroll
    for (int i = 0; i < 4; ++i) { float d = v[i] - m; var += d * d; }
    #pragma unroll
    for (int off = 16; off > 0; off >>= 1) var += __shfl_xor(var, off, 32);
    var *= (1.0f / 128.0f);
    float r = rsqrtf(var + 1e-5f);
    #pragma unroll
    for (int i = 0; i < 4; ++i) {
        int c = lane + 32 * i;
        out[base + c] = (v[i] - m) * r * s[c] + bi[c];
    }
}

__global__ void citing_cat(const float* __restrict__ x2, const float* __restrict__ nf,
                           const int* __restrict__ citing, float* __restrict__ cp) {
    int b = blockIdx.x, d = threadIdx.x;   // blockDim = 128
    float sm = 0.0f;
    #pragma unroll
    for (int l = 0; l < 20; ++l) sm += x2[((long)b * 20 + l) * 128 + d];
    cp[(long)b * 256 + d]       = sm * (1.0f / 20.0f);
    cp[(long)b * 256 + 128 + d] = nf[(long)citing[b] * 128 + d];
}

// ---------------------------------------------------------------------------
// Host-side launcher
// ---------------------------------------------------------------------------
template<int EPI, bool XT, int WC>
static void launch_gemm(dim3 grid, hipStream_t st,
                        const float* A, long lda, long aZ,
                        const float* X, long ldx, long xZ,
                        float* C, long ldc, long cZ,
                        const float* bias, int M, int K, int Nc, float sc) {
    gemm_wmma<EPI, XT, WC><<<grid, dim3(256), 0, st>>>(
        A, lda, aZ, X, ldx, xZ, C, ldc, cZ, bias, M, K, Nc, sc);
}

extern "C" void kernel_launch(void* const* d_in, const int* in_sizes, int n_in,
                              void* d_out, int out_size, void* d_ws, size_t ws_size,
                              hipStream_t stream) {
    (void)in_sizes; (void)n_in; (void)out_size; (void)ws_size;

    const int N = 12000, D = 128, B = 512, L = 20, BL = B * L, F = 400, FF = 512;

    const float* emb   = (const float*)d_in[0];
    const float* f1w   = (const float*)d_in[1];
    const float* f1b   = (const float*)d_in[2];
    const float* g0w1  = (const float*)d_in[3];
    const float* g0b1  = (const float*)d_in[4];
    const float* g0w2  = (const float*)d_in[5];
    const float* g0b2  = (const float*)d_in[6];
    const float* g1w1  = (const float*)d_in[7];
    const float* g1b1  = (const float*)d_in[8];
    const float* g1w2  = (const float*)d_in[9];
    const float* g1b2  = (const float*)d_in[10];
    const float* ipw   = (const float*)d_in[11];
    const float* ipb   = (const float*)d_in[12];
    const float* opw   = (const float*)d_in[13];
    const float* opb   = (const float*)d_in[14];
    const float* ln1s  = (const float*)d_in[15];
    const float* ln1b  = (const float*)d_in[16];
    const float* ff1w  = (const float*)d_in[17];
    const float* ff1b  = (const float*)d_in[18];
    const float* ff2w  = (const float*)d_in[19];
    const float* ff2b  = (const float*)d_in[20];
    const float* ln2s  = (const float*)d_in[21];
    const float* ln2b  = (const float*)d_in[22];
    const float* adj   = (const float*)d_in[23];
    const float* nfeat = (const float*)d_in[24];
    const int*   citing = (const int*)d_in[26];
    const int*   useqs  = (const int*)d_in[28];
    float* scores = (float*)d_out;

    float* ws  = (float*)d_ws;
    float* nfB = ws;  ws += (long)N * D;
    float* xB  = ws;  ws += (long)N * D;
    float* n1  = ws;  ws += (long)N * 256;
    float* n2  = ws;  ws += (long)N * 256;
    float* seq = ws;  ws += (long)BL * D;
    float* qkv = ws;  ws += (long)BL * 384;
    float* Sb  = ws;  ws += (long)4 * 512 * 512;
    float* oAt = ws;  ws += (long)BL * D;
    float* oPr = ws;  ws += (long)BL * D;
    float* x1  = ws;  ws += (long)BL * D;
    float* ffh = ws;  ws += (long)BL * FF;
    float* ffo = ws;  ws += (long)BL * D;
    float* x2  = ws;  ws += (long)BL * D;
    float* cp  = ws;  ws += (long)B * 256;
    float* cit = ws;  ws += (long)N * 256;

    const dim3 gN1(188, 1, 1);
    const dim3 gBL1(160, 1, 1);

    // nf = node_feature @ f1_w^T + f1_b
    launch_gemm<EPI_BIAS, true, 16>(gN1, stream, nfeat, F, 0, f1w, F, 0,
                                    nfB, D, 0, f1b, N, F, D, 0.f);
    // x0 = embedding * nf
    ew_mul<<<dim3((N * D + 255) / 256), dim3(256), 0, stream>>>(emb, nfB, xB, N * D);

    // GCN module 0
    launch_gemm<EPI_NONE,      false, 16>(gN1, stream, adj, N, 0, xB, D, 0,
                                          n1, D, 0, nullptr, N, N, D, 0.f);
    launch_gemm<EPI_BIAS_RELU, false, 32>(gN1, stream, n1, D, 0, g0w1, 256, 0,
                                          n2, 256, 0, g0b1, N, D, 256, 0.f);
    launch_gemm<EPI_NONE,      false, 32>(gN1, stream, adj, N, 0, n2, 256, 0,
                                          n1, 256, 0, nullptr, N, N, 256, 0.f);
    launch_gemm<EPI_BIAS,      false, 16>(gN1, stream, n1, 256, 0, g0w2, D, 0,
                                          xB, D, 0, g0b2, N, 256, D, 0.f);
    // GCN module 1
    launch_gemm<EPI_NONE,      false, 16>(gN1, stream, adj, N, 0, xB, D, 0,
                                          n1, D, 0, nullptr, N, N, D, 0.f);
    launch_gemm<EPI_BIAS_RELU, false, 32>(gN1, stream, n1, D, 0, g1w1, 256, 0,
                                          n2, 256, 0, g1b1, N, D, 256, 0.f);
    launch_gemm<EPI_NONE,      false, 32>(gN1, stream, adj, N, 0, n2, 256, 0,
                                          n1, 256, 0, nullptr, N, N, 256, 0.f);
    launch_gemm<EPI_BIAS,      false, 16>(gN1, stream, n1, 256, 0, g1w2, D, 0,
                                          xB, D, 0, g1b2, N, 256, D, 0.f);

    // seq = item[u_seqs]
    gather_rows128<<<dim3((BL * D + 255) / 256), dim3(256), 0, stream>>>(xB, useqs, seq, BL);

    // qkv = seq @ in_proj_w^T + in_proj_b
    launch_gemm<EPI_BIAS, true, 16>(dim3(160, 3, 1), stream, seq, D, 0, ipw, D, 0,
                                    qkv, 384, 0, ipb, BL, D, 384, 0.f);

    // attention, batched over heads (grid.z=4), looped over l
    const float scale = 0.17677669529663687f;   // 1/sqrt(32)
    for (int l = 0; l < L; ++l) {
        const float* qp = qkv + (long)l * 384;
        const float* kp = qkv + (long)l * 384 + 128;
        const float* vp = qkv + (long)l * 384 + 256;
        launch_gemm<EPI_SCALE, true, 16>(dim3(8, 4, 4), stream,
                                         qp, 384L * L, 32, kp, 384L * L, 32,
                                         Sb, 512, 512L * 512, nullptr,
                                         512, 32, 512, scale);
        softmax512<<<dim3(256), dim3(256), 0, stream>>>(Sb, 4 * 512);
        launch_gemm<EPI_NONE, false, 16>(dim3(8, 1, 4), stream,
                                         Sb, 512, 512L * 512, vp, 384L * L, 32,
                                         oAt + (long)l * 128, (long)L * 128, 32, nullptr,
                                         512, 512, 32, 0.f);
    }

    // out_proj
    launch_gemm<EPI_BIAS, true, 16>(gBL1, stream, oAt, D, 0, opw, D, 0,
                                    oPr, D, 0, opb, BL, D, D, 0.f);
    // x1 = LN(seq + oPr)
    add_ln128<<<dim3((BL + 7) / 8), dim3(256), 0, stream>>>(seq, oPr, ln1s, ln1b, x1, BL);
    // ff = relu(x1 @ ff1_w^T + b) @ ff2_w^T + b
    launch_gemm<EPI_BIAS_RELU, true, 32>(dim3(160, 2, 1), stream, x1, D, 0, ff1w, D, 0,
                                         ffh, FF, 0, ff1b, BL, D, FF, 0.f);
    launch_gemm<EPI_BIAS, true, 16>(gBL1, stream, ffh, FF, 0, ff2w, FF, 0,
                                    ffo, D, 0, ff2b, BL, FF, D, 0.f);
    // x2 = LN(x1 + ff)
    add_ln128<<<dim3((BL + 7) / 8), dim3(256), 0, stream>>>(x1, ffo, ln2s, ln2b, x2, BL);

    // citing_paper / cited_paper
    citing_cat<<<dim3(B), dim3(128), 0, stream>>>(x2, nfB, citing, cp);
    concat_cited_async<<<dim3((N + 15) / 16), dim3(256), 0, stream>>>(xB, nfB, cit, N);

    // scores = citing_paper @ cited_paper^T
    launch_gemm<EPI_NONE, true, 16>(dim3(8, 94, 1), stream, cp, 256, 0, cit, 256, 0,
                                    scores, N, 0, nullptr, B, 256, N, 0.f);
}
